// GaussianRenderer_48584670052373
// MI455X (gfx1250) — matrix operand order repaired and checked by reading, not compiled, not run on hardware
//
#include <hip/hip_runtime.h>
#include <hip/hip_bf16.h>
#include <stdint.h>

typedef __attribute__((ext_vector_type(16))) __bf16 v16bf;
typedef __attribute__((ext_vector_type(8)))  float  v8f;

static constexpr int kW      = 200;
static constexpr int kNPIX   = 40000;   // 200*200
static constexpr int kP      = 4096;    // gaussians per batch
static constexpr int kB      = 2;
static constexpr int kD      = 32;
static constexpr float kTHR  = 0.05f;
static constexpr float kAMIN = 1.0f / 255.0f;
static constexpr float kL2E  = 1.4426950408889634f;

// ---------------- preprocessing: per-gaussian conic params -----------------
// Stores exp2-ready quadratic coefficients: e(dx,dy) = a2*dx^2 + b2*dx*dy + c2*dy^2
// with a2 = -0.5*log2e*conicA, b2 = -log2e*conicB, c2 = -0.5*log2e*conicC,
// so alpha = op * exp2(e) and (power > 0) <=> (e > 0).
__global__ void prep_params(const float* __restrict__ means,
                            const float* __restrict__ cov6,
                            const float* __restrict__ opac,
                            float* __restrict__ params) {
  int i = blockIdx.x * blockDim.x + threadIdx.x;
  if (i >= kB * kP) return;
  float mx = means[(size_t)i * 3 + 0];
  float my = means[(size_t)i * 3 + 1];
  float px = -2.0f * my + 100.0f;   // image x
  float py = -2.0f * mx + 100.0f;   // image y
  float a  = 4.0f * cov6[(size_t)i * 6 + 3] + 0.3f;
  float bb = 4.0f * cov6[(size_t)i * 6 + 1];
  float c  = 4.0f * cov6[(size_t)i * 6 + 0] + 0.3f;
  float det = a * c - bb * bb;
  float inv = det > 0.0f ? 1.0f / det : 0.0f;
  float o = opac[i];
  o = (o > kTHR && det > 0.0f) ? o : 0.0f;
  float* p = params + (size_t)i * 8;
  p[0] = px;
  p[1] = py;
  p[2] = -0.5f * kL2E * (c * inv);   // a2 (dx^2 coeff)
  p[3] =  kL2E * (bb * inv);         // b2 (dx*dy coeff) = -log2e * (-bb*inv)
  p[4] = -0.5f * kL2E * (a * inv);   // c2 (dy^2 coeff)
  p[5] = o; p[6] = 0.0f; p[7] = 0.0f;
}

// -------------- preprocessing: features -> bf16, channel-major -------------
__global__ void prep_feat(const float* __restrict__ feats,
                          uint16_t* __restrict__ featT) {
  int i = blockIdx.x * blockDim.x + threadIdx.x;
  if (i >= kB * kD * kP) return;
  int g = i & (kP - 1);
  int m = (i >> 12) & (kD - 1);
  int b = i >> 17;
  float v = feats[((size_t)(b * kP + g)) * kD + m];
  uint32_t u = __float_as_uint(v);
  uint32_t r = u + 0x7FFFu + ((u >> 16) & 1u);   // round-to-nearest-even
  featT[((size_t)(b * kD + m)) * kP + g] = (uint16_t)(r >> 16);
}

// ---------------------------- main renderer --------------------------------
// One v_perm_b32: packed = {hi.b3, hi.b2, lo.b3, lo.b2} (bf16 truncation pack)
__device__ __forceinline__ uint32_t pack_bf16(float lo, float hi) {
  return __builtin_amdgcn_perm(__float_as_uint(hi), __float_as_uint(lo),
                               0x07060302u);
}

// store->load RAW through LDS: keep the dscnt wait (hazard insurance)
__device__ __forceinline__ void wave_lds_sync() {
  __builtin_amdgcn_wave_barrier();
  asm volatile("s_wait_dscnt 0x0" ::: "memory");
  __builtin_amdgcn_wave_barrier();
}

// compiler-only ordering fence (per-wave DS ops are hardware in-order)
__device__ __forceinline__ void wave_order_fence() {
  asm volatile("" ::: "memory");
}

union BFrag { uint4 q[2]; v16bf v; };

__global__ void __launch_bounds__(256)
render(const float* __restrict__ params, const uint16_t* __restrict__ featT,
       float* __restrict__ out) {
  // per-wave private LDS regions (no block barriers needed)
  __shared__ __align__(16) float    sparams[8][32 * 8];   //  8 KB
  __shared__ __align__(16) uint32_t swgt[8][32 * 16];     // 16 KB

  const int wave = threadIdx.x >> 5;
  const int lane = threadIdx.x & 31;
  const int b    = blockIdx.y;
  const int p0   = blockIdx.x * 256 + wave * 32;
  if (p0 >= kNPIX) return;                 // wave-uniform early exit

  int myPix = p0 + lane;
  if (myPix > kNPIX - 1) myPix = kNPIX - 1;  // clamp; stores are guarded
  const float fx = (float)(myPix % kW);
  const float fy = (float)(myPix / kW);
  const int n    = lane & 15;
  const int half = lane >> 4;

  float T = 1.0f;                          // per-pixel transmittance
  v8f c00 = {}, c01 = {}, c10 = {}, c11 = {};

  const float4*   gparams = (const float4*)(params + (size_t)b * kP * 8);
  const uint16_t* fbase0  = featT + ((size_t)b * kD + n) * kP;       // ch 0..15
  const uint16_t* fbase1  = fbase0 + (size_t)16 * kP;                // ch 16..31
  float4*   sp = (float4*)&sparams[wave][0];
  uint32_t* wb = &swgt[wave][0];

#pragma unroll 1
  for (int chunk = 0; chunk < kP / 32; ++chunk) {
    const int gbase = chunk * 32;

    // A fragments first: latency hides under the serial alpha loop below.
    // 16-bit A layout: lanes 0-15 elems = K{0..7,16..23}; lanes 16-31 = K{8..15,24..31}
    BFrag a0, a1;
    a0.q[0] = *(const uint4*)(fbase0 + gbase + half * 8);
    a0.q[1] = *(const uint4*)(fbase0 + gbase + 16 + half * 8);
    a1.q[0] = *(const uint4*)(fbase1 + gbase + half * 8);
    a1.q[1] = *(const uint4*)(fbase1 + gbase + 16 + half * 8);

    // stage this chunk's 32 gaussian params into LDS (2x b128 per lane)
    sp[lane * 2 + 0] = gparams[(size_t)(gbase + lane) * 2 + 0];
    sp[lane * 2 + 1] = gparams[(size_t)(gbase + lane) * 2 + 1];
    wave_lds_sync();

    // sequential front-to-back alpha over 32 gaussians for this lane's pixel
    uint32_t wpack[16];
#pragma unroll
    for (int g2 = 0; g2 < 16; ++g2) {
      float wlo, whi;
#pragma unroll
      for (int k = 0; k < 2; ++k) {
        const int g = g2 * 2 + k;
        const float4 q0 = sp[g * 2 + 0];   // px, py, a2, b2  (LDS broadcast)
        const float4 q1 = sp[g * 2 + 1];   // c2, op, -, -
        const float dx = q0.x - fx;
        const float dy = q0.y - fy;
        const float u  = fmaf(q0.z, dx, q0.w * dy);       // a2*dx + b2*dy
        const float e  = fmaf(q1.x * dy, dy, u * dx);     // + c2*dy^2
        float al = fminf(q1.y * exp2f(e), 0.99f);
        al = (e > 0.0f) ? 0.0f : al;
        al = (al < kAMIN) ? 0.0f : al;
        const float wgt = al * T;
        T = fmaf(-al, T, T);
        if (k == 0) wlo = wgt; else whi = wgt;
      }
      wpack[g2] = pack_bf16(wlo, whi);
    }

    // write 32 bf16 weights for this pixel (row-major by pixel, 64B/row)
    {
      uint4* wrow = (uint4*)&wb[lane * 16];
      wrow[0] = make_uint4(wpack[0], wpack[1], wpack[2], wpack[3]);
      wrow[1] = make_uint4(wpack[4], wpack[5], wpack[6], wpack[7]);
      wrow[2] = make_uint4(wpack[8], wpack[9], wpack[10], wpack[11]);
      wrow[3] = make_uint4(wpack[12], wpack[13], wpack[14], wpack[15]);
    }
    wave_lds_sync();

    // B fragments: weights, K=32 gaussians x N=16 pixels (two N-blocks)
    BFrag b0, b1;
    {
      const uint4* r0 = (const uint4*)&wb[(n)*16 + half * 4];
      const uint4* r1 = (const uint4*)&wb[(n + 16) * 16 + half * 4];
      b0.q[0] = r0[0]; b0.q[1] = r0[2];
      b1.q[0] = r1[0]; b1.q[1] = r1[2];
    }

    c00 = __builtin_amdgcn_wmma_f32_16x16x32_bf16(false, a0.v, false, b0.v,
                                                  (short)0, c00, false, false);
    c01 = __builtin_amdgcn_wmma_f32_16x16x32_bf16(false, a0.v, false, b1.v,
                                                  (short)0, c01, false, false);
    c10 = __builtin_amdgcn_wmma_f32_16x16x32_bf16(false, a1.v, false, b0.v,
                                                  (short)0, c10, false, false);
    c11 = __builtin_amdgcn_wmma_f32_16x16x32_bf16(false, a1.v, false, b1.v,
                                                  (short)0, c11, false, false);
    wave_order_fence();   // next chunk's LDS stores stay behind these reads
  }

  // D layout (32-bit 16x16): lanes 0-15 -> N=lane, M=r; lanes 16-31 -> N=lane-16, M=8+r
  const int mBase = half * 8;
#pragma unroll
  for (int r = 0; r < 8; ++r) {
    const int pixA = p0 + n;
    const int pixB = p0 + 16 + n;
    const int mA = mBase + r;
    const int mB = 16 + mBase + r;
    if (pixA < kNPIX) {
      out[((size_t)b * kD + mA) * kNPIX + pixA] = c00[r];
      out[((size_t)b * kD + mB) * kNPIX + pixA] = c10[r];
    }
    if (pixB < kNPIX) {
      out[((size_t)b * kD + mA) * kNPIX + pixB] = c01[r];
      out[((size_t)b * kD + mB) * kNPIX + pixB] = c11[r];
    }
  }
}

// --------------------------- count reduction -------------------------------
__global__ void count_kernel(const float* __restrict__ opac,
                             float* __restrict__ out) {
  __shared__ float red[256];
  float s = 0.0f;
  for (int i = threadIdx.x; i < kB * kP; i += 256)
    s += (opac[i] > kTHR) ? 1.0f : 0.0f;
  red[threadIdx.x] = s;
  __syncthreads();
  for (int off = 128; off > 0; off >>= 1) {
    if (threadIdx.x < off) red[threadIdx.x] += red[threadIdx.x + off];
    __syncthreads();
  }
  if (threadIdx.x == 0) out[0] = red[0] * (1.0f / 128.0f);
}

// ------------------------------- launcher ----------------------------------
extern "C" void kernel_launch(void* const* d_in, const int* in_sizes, int n_in,
                              void* d_out, int out_size, void* d_ws, size_t ws_size,
                              hipStream_t stream) {
  const float* feats = (const float*)d_in[0];
  const float* means = (const float*)d_in[1];
  const float* cov6  = (const float*)d_in[2];
  const float* opac  = (const float*)d_in[3];
  float* out = (float*)d_out;

  float*    params = (float*)d_ws;                                       // 256 KB
  uint16_t* featT  = (uint16_t*)((char*)d_ws +
                                 (size_t)kB * kP * 8 * sizeof(float));   // 512 KB

  prep_params<<<(kB * kP + 255) / 256, 256, 0, stream>>>(means, cov6, opac, params);
  prep_feat<<<(kB * kD * kP + 255) / 256, 256, 0, stream>>>(feats, featT);

  dim3 grid((kNPIX + 255) / 256, kB);
  render<<<grid, 256, 0, stream>>>(params, featT, out);

  count_kernel<<<1, 256, 0, stream>>>(opac, out + (size_t)kB * kD * kNPIX);
}
